// CrossAttention_31705448580001
// MI455X (gfx1250) — matrix operand order repaired and checked by reading, not compiled, not run on hardware
//
#include <hip/hip_runtime.h>
#include <hip/hip_bf16.h>

// ---------------------------------------------------------------------------
// CDNA5 (gfx1250) cross-attention, bf16 WMMA path.
// Sizes (fixed by reference): B=2, N=2048, M=4096, QD=1024, CD=768,
// H=16, DH=64, INNER=1024.
// ---------------------------------------------------------------------------

typedef __attribute__((ext_vector_type(16))) __bf16 v16bf;
typedef __attribute__((ext_vector_type(8)))  float  v8f;

#define LOG2E 1.44269504088896340736f

__device__ __forceinline__ v8f wmma_bf16(v16bf a, v16bf b, v8f c) {
    return __builtin_amdgcn_wmma_f32_16x16x32_bf16(
        /*neg_a=*/false, a, /*neg_b=*/false, b,
        /*c_mod=*/(short)0, c, /*reuse_a=*/false, /*reuse_b=*/false);
}

// A fragment (16x32, 16-bit): lane L -> row L%16;
// elements 0..7 = K[(L/16)*8 .. +7], elements 8..15 = K[(L/16)*8+16 .. +7].
__device__ __forceinline__ v16bf load_frag_a(const __bf16* p, int ld) {
    const int lane = threadIdx.x & 31;
    const int row  = lane & 15;
    const int kb   = (lane >> 4) << 3;
    const __bf16* q = p + (size_t)row * ld + kb;
    union { v16bf v; uint4 u[2]; } t;
    t.u[0] = *(const uint4*)(q);
    t.u[1] = *(const uint4*)(q + 16);
    return t.v;
}

// B fragment (32x16, 16-bit), from a row-major (N x K) "B-transposed" buffer:
// lane L -> column L%16; elements 0..7 = K[(L/16)*8 .. +7], elements 8..15 = K[(L/16)*8+16 .. +7].
__device__ __forceinline__ v16bf load_frag_b(const __bf16* p, int ld) {
    const int lane = threadIdx.x & 31;
    const int col  = lane & 15;
    const int kb   = (lane >> 4) << 3;
    const __bf16* q = p + (size_t)col * ld + kb;
    union { v16bf v; uint4 u[2]; } t;
    t.u[0] = *(const uint4*)(q);
    t.u[1] = *(const uint4*)(q + 16);
    return t.v;
}

__device__ __forceinline__ float red_max16(float v) {
    v = fmaxf(v, __shfl_xor(v, 1, 32));
    v = fmaxf(v, __shfl_xor(v, 2, 32));
    v = fmaxf(v, __shfl_xor(v, 4, 32));
    v = fmaxf(v, __shfl_xor(v, 8, 32));
    return v;
}
__device__ __forceinline__ float red_sum16(float v) {
    v += __shfl_xor(v, 1, 32);
    v += __shfl_xor(v, 2, 32);
    v += __shfl_xor(v, 4, 32);
    v += __shfl_xor(v, 8, 32);
    return v;
}

// --------------------------- conversion kernels ----------------------------

__global__ void ca_to_bf16(const float* __restrict__ src,
                           __bf16* __restrict__ dst, int n) {
    int i = blockIdx.x * blockDim.x + threadIdx.x;
    if (i < n) dst[i] = (__bf16)src[i];
}

// src: K x N (row major f32) -> dst: N x K (row major bf16)
__global__ void ca_transpose_bf16(const float* __restrict__ src,
                                  __bf16* __restrict__ dst, int K, int N) {
    int i = blockIdx.x * blockDim.x + threadIdx.x;
    if (i < K * N) {
        int k = i / N, n = i % N;
        dst[(size_t)n * K + k] = (__bf16)src[i];
    }
}

// ------------------------------- WMMA GEMM ---------------------------------
// C[M x N] = alpha * (A[M x K] @ Bt[N x K]^T) (+ bias[col]).
// Block = 256 threads = 8 waves laid out 4(M) x 2(N); wave tile 32x32;
// block tile 128(M) x 64(N). M%128==0, N%64==0, K%32==0.

template <bool F32OUT, bool BIAS>
__global__ __launch_bounds__(256) void ca_gemm_wmma(
    const __bf16* __restrict__ A, const __bf16* __restrict__ Bt,
    void* __restrict__ Cout, const float* __restrict__ bias,
    int M, int N, int K, float alpha) {
    const int lane = threadIdx.x & 31;
    const int w    = threadIdx.x >> 5;
    const int wm   = w & 3;
    const int wn   = w >> 2;
    const int m0   = blockIdx.x * 128 + wm * 32;
    const int n0   = blockIdx.y * 64 + wn * 32;

    v8f acc[2][2] = {};
    for (int k0 = 0; k0 < K; k0 += 32) {
        v16bf a0 = load_frag_a(A + (size_t)m0 * K + k0, K);
        v16bf a1 = load_frag_a(A + (size_t)(m0 + 16) * K + k0, K);
        v16bf b0 = load_frag_b(Bt + (size_t)n0 * K + k0, K);
        v16bf b1 = load_frag_b(Bt + (size_t)(n0 + 16) * K + k0, K);
        acc[0][0] = wmma_bf16(a0, b0, acc[0][0]);
        acc[0][1] = wmma_bf16(a0, b1, acc[0][1]);
        acc[1][0] = wmma_bf16(a1, b0, acc[1][0]);
        acc[1][1] = wmma_bf16(a1, b1, acc[1][1]);
    }

    const int crow = (lane >> 4) << 3;   // row base within 16-row frag
    const int ccol = lane & 15;
#pragma unroll
    for (int i = 0; i < 2; ++i)
#pragma unroll
        for (int j = 0; j < 2; ++j)
#pragma unroll
            for (int e = 0; e < 8; ++e) {
                int r = m0 + i * 16 + crow + e;
                int c = n0 + j * 16 + ccol;
                float v = acc[i][j][e] * alpha;
                if (BIAS) v += bias[c];
                if (F32OUT) ((float*)Cout)[(size_t)r * N + c] = v;
                else        ((__bf16*)Cout)[(size_t)r * N + c] = (__bf16)v;
            }
}

// ---------------------------- flash attention ------------------------------
// Q,K,V: bf16 (B*rows x 1024), per (b,h) slice has leading dim 1024, offset
// h*64. Scale 1/sqrt(64) was folded into Q. One block = 8 waves, each wave
// owns 32 query rows; loop over ctx (M=4096) in chunks of 32 with online
// softmax. V chunk staged transposed in LDS; P staged per-wave in LDS so it
// can be re-read in WMMA A-fragment layout.

__global__ __launch_bounds__(256) void ca_flash_attn(
    const __bf16* __restrict__ Q, const __bf16* __restrict__ Kc,
    const __bf16* __restrict__ Vc, const unsigned char* __restrict__ mask,
    __bf16* __restrict__ O) {
    constexpr int NQ = 2048, MC = 4096, INNER = 1024, DH = 64;

    __shared__ __align__(16) __bf16 lds_vt[DH * 32];      // V^T tile: 64 x 32
    __shared__ __align__(16) __bf16 lds_p[8][32 * 32];    // per-wave P tile

    const int lane  = threadIdx.x & 31;
    const int w     = threadIdx.x >> 5;
    const int tile  = blockIdx.x & 7;        // 8 row-tiles of 256 per (b,h)
    const int bh    = blockIdx.x >> 3;
    const int b     = bh >> 4;
    const int h     = bh & 15;
    const int qrow0 = tile * 256 + w * 32;

    const __bf16* qbase = Q + ((size_t)(b * NQ + qrow0)) * INNER + h * DH;
    const __bf16* kbase = Kc + ((size_t)b * MC) * INNER + h * DH;
    const __bf16* vbase = Vc + ((size_t)b * MC) * INNER + h * DH;
    const unsigned char* mbase = mask + (size_t)b * MC;

    v16bf qa[2][2];
#pragma unroll
    for (int rm = 0; rm < 2; ++rm)
#pragma unroll
        for (int kc = 0; kc < 2; ++kc)
            qa[rm][kc] = load_frag_a(qbase + (size_t)rm * 16 * INNER + kc * 32, INNER);

    v8f oacc[2][4] = {};
    float mst[2][8], lst[2][8];
#pragma unroll
    for (int rm = 0; rm < 2; ++rm)
#pragma unroll
        for (int e = 0; e < 8; ++e) { mst[rm][e] = -3.0e38f; lst[rm][e] = 0.0f; }

    const int crow = (lane >> 4) << 3;
    const int ccol = lane & 15;

    for (int c0 = 0; c0 < MC; c0 += 32) {
        __syncthreads();  // previous chunk's readers of lds_vt are done
        {   // cooperative transposed load of V chunk: (32 ctx x 64 dim) -> vt
            int e = threadIdx.x * 8;
            int c = e >> 6;        // ctx within chunk
            int d = e & 63;        // dim start (8 contiguous dims)
            union { uint4 u; __bf16 hv[8]; } t;
            t.u = *(const uint4*)(vbase + (size_t)(c0 + c) * INNER + d);
#pragma unroll
            for (int q8 = 0; q8 < 8; ++q8) lds_vt[(d + q8) * 32 + c] = t.hv[q8];
        }
        __syncthreads();

        const float mk0 = mbase[c0 + ccol]      ? 0.0f : -3.0e38f;
        const float mk1 = mbase[c0 + 16 + ccol] ? 0.0f : -3.0e38f;

        // ---- S = Q @ K_chunk^T (32 x 32) ----
        v8f s[2][2] = {};
        v16bf kb[2][2];
#pragma unroll
        for (int cn = 0; cn < 2; ++cn)
#pragma unroll
            for (int kc = 0; kc < 2; ++kc)
                kb[cn][kc] = load_frag_b(
                    kbase + (size_t)(c0 + cn * 16) * INNER + kc * 32, INNER);
#pragma unroll
        for (int rm = 0; rm < 2; ++rm)
#pragma unroll
            for (int cn = 0; cn < 2; ++cn) {
                s[rm][cn] = wmma_bf16(qa[rm][0], kb[cn][0], s[rm][cn]);
                s[rm][cn] = wmma_bf16(qa[rm][1], kb[cn][1], s[rm][cn]);
            }

        // ---- online softmax + stage P in LDS ----
#pragma unroll
        for (int rm = 0; rm < 2; ++rm) {
#pragma unroll
            for (int e = 0; e < 8; ++e) {
                float s0 = s[rm][0][e] + mk0;
                float s1 = s[rm][1][e] + mk1;
                float rmx = red_max16(fmaxf(s0, s1));
                float mo  = mst[rm][e];
                float mn  = fmaxf(mo, rmx);
                float sc  = exp2f((mo - mn) * LOG2E);
                float p0  = exp2f((s0 - mn) * LOG2E);
                float p1  = exp2f((s1 - mn) * LOG2E);
                float rs  = red_sum16(p0 + p1);
                mst[rm][e] = mn;
                lst[rm][e] = lst[rm][e] * sc + rs;
#pragma unroll
                for (int j = 0; j < 4; ++j) oacc[rm][j][e] *= sc;
                int r = rm * 16 + crow + e;
                lds_p[w][r * 32 + ccol]      = (__bf16)p0;
                lds_p[w][r * 32 + 16 + ccol] = (__bf16)p1;
            }
        }
        __syncthreads();

        // ---- O += P @ V_chunk (A-frags from lds_p, B-frags from lds_vt) ----
        v16bf pa0 = load_frag_a(&lds_p[w][0], 32);
        v16bf pa1 = load_frag_a(&lds_p[w][16 * 32], 32);
#pragma unroll
        for (int j = 0; j < 4; ++j) {
            v16bf vb = load_frag_b(&lds_vt[(j * 16) * 32], 32);
            oacc[0][j] = wmma_bf16(pa0, vb, oacc[0][j]);
            oacc[1][j] = wmma_bf16(pa1, vb, oacc[1][j]);
        }
    }

    // ---- epilogue: O /= l, store bf16 ----
#pragma unroll
    for (int rm = 0; rm < 2; ++rm)
#pragma unroll
        for (int e = 0; e < 8; ++e) {
            float inv = 1.0f / lst[rm][e];
            int r = qrow0 + rm * 16 + crow + e;
#pragma unroll
            for (int j = 0; j < 4; ++j) {
                int col = h * DH + j * 16 + ccol;
                O[((size_t)(b * NQ + r)) * INNER + col] =
                    (__bf16)(oacc[rm][j][e] * inv);
            }
        }
}

// ------------------------------- launcher ----------------------------------

extern "C" void kernel_launch(void* const* d_in, const int* in_sizes, int n_in,
                              void* d_out, int out_size, void* d_ws, size_t ws_size,
                              hipStream_t stream) {
    (void)in_sizes; (void)n_in; (void)out_size; (void)ws_size;
    constexpr int B = 2, NQ = 2048, MC = 4096, QD = 1024, CD = 768, INNER = 1024;

    const float* x    = (const float*)d_in[0];
    const float* ctx  = (const float*)d_in[1];
    const unsigned char* cmask = (const unsigned char*)d_in[2];
    const float* Wq   = (const float*)d_in[3];
    const float* Wk   = (const float*)d_in[4];
    const float* Wv   = (const float*)d_in[5];
    const float* Wo   = (const float*)d_in[6];
    const float* bo   = (const float*)d_in[7];
    float* out        = (float*)d_out;

    // Workspace carve-up (bf16 buffers, 256B aligned).
    char* ws = (char*)d_ws;
    auto carve = [&](size_t elems) {
        void* p = (void*)ws;
        ws += (elems * sizeof(__bf16) + 255) & ~(size_t)255;
        return (__bf16*)p;
    };
    __bf16* xb  = carve((size_t)B * NQ * QD);      // x in bf16
    __bf16* cb  = carve((size_t)B * MC * CD);      // context in bf16
    __bf16* wqt = carve((size_t)INNER * QD);       // Wq^T
    __bf16* wkt = carve((size_t)INNER * CD);       // Wk^T
    __bf16* wvt = carve((size_t)INNER * CD);       // Wv^T
    __bf16* wot = carve((size_t)QD * INNER);       // Wo^T
    __bf16* qb  = carve((size_t)B * NQ * INNER);   // Q (scaled)
    __bf16* kb  = carve((size_t)B * MC * INNER);   // K
    __bf16* vb  = carve((size_t)B * MC * INNER);   // V
    __bf16* ao  = carve((size_t)B * NQ * INNER);   // attention output

    const int T = 256;
    ca_to_bf16<<<(B * NQ * QD + T - 1) / T, T, 0, stream>>>(x, xb, B * NQ * QD);
    ca_to_bf16<<<(B * MC * CD + T - 1) / T, T, 0, stream>>>(ctx, cb, B * MC * CD);
    ca_transpose_bf16<<<(QD * INNER + T - 1) / T, T, 0, stream>>>(Wq, wqt, QD, INNER);
    ca_transpose_bf16<<<(CD * INNER + T - 1) / T, T, 0, stream>>>(Wk, wkt, CD, INNER);
    ca_transpose_bf16<<<(CD * INNER + T - 1) / T, T, 0, stream>>>(Wv, wvt, CD, INNER);
    ca_transpose_bf16<<<(INNER * QD + T - 1) / T, T, 0, stream>>>(Wo, wot, INNER, QD);

    // Projections (bf16 WMMA, f32 accumulate). Softmax scale folded into Q.
    ca_gemm_wmma<false, false><<<dim3((B * NQ) / 128, INNER / 64), T, 0, stream>>>(
        xb, wqt, qb, nullptr, B * NQ, INNER, QD, 0.125f);
    ca_gemm_wmma<false, false><<<dim3((B * MC) / 128, INNER / 64), T, 0, stream>>>(
        cb, wkt, kb, nullptr, B * MC, INNER, CD, 1.0f);
    ca_gemm_wmma<false, false><<<dim3((B * MC) / 128, INNER / 64), T, 0, stream>>>(
        cb, wvt, vb, nullptr, B * MC, INNER, CD, 1.0f);

    // Flash attention: 8 row-tiles of 256 per (b,h) -> 2*16*8 = 256 blocks.
    ca_flash_attn<<<B * 16 * (NQ / 256), T, 0, stream>>>(qb, kb, vb, cmask, ao);

    // Output projection with bias, f32 output.
    ca_gemm_wmma<true, true><<<dim3((B * NQ) / 128, QD / 64), T, 0, stream>>>(
        ao, wot, out, bo, B * NQ, QD, INNER, 1.0f);
}
